// TernaryLinear_8942121910926
// MI455X (gfx1250) — matrix operand order repaired and checked by reading, not compiled, not run on hardware
//
#include <hip/hip_runtime.h>

typedef __attribute__((ext_vector_type(16))) _Float16 v16h;
typedef __attribute__((ext_vector_type(8)))  _Float16 v8h;
typedef __attribute__((ext_vector_type(8)))  float    v8f;

#define D_IN_  4096
#define D_OUT_ 4096
#define N_TOK_ 8192

// ---------------- pack kernels ----------------

__global__ void pack_x_f16(const float* __restrict__ x, _Float16* __restrict__ xh, int n4) {
    int i = blockIdx.x * blockDim.x + threadIdx.x;
    if (i >= n4) return;
    const float4 v = ((const float4*)x)[i];
    v8h* dst = (v8h*)xh;          // write 4 halfs = 8B; pack two threads' worth? no: use half4 via struct
    _Float16 h0 = (_Float16)v.x, h1 = (_Float16)v.y, h2 = (_Float16)v.z, h3 = (_Float16)v.w;
    // 8-byte store
    struct __attribute__((packed)) H4 { _Float16 a,b,c,d; };
    ((H4*)xh)[i] = H4{h0,h1,h2,h3};
    (void)dst;
}

__device__ __forceinline__ float signf(float w) {
    return (float)((w > 0.0f) - (w < 0.0f));
}

__global__ void pack_wdiff_f16(const float* __restrict__ W1, const float* __restrict__ W2,
                               _Float16* __restrict__ wd, int n4) {
    int i = blockIdx.x * blockDim.x + threadIdx.x;
    if (i >= n4) return;
    const float4 a = ((const float4*)W1)[i];
    const float4 b = ((const float4*)W2)[i];
    struct __attribute__((packed)) H4 { _Float16 a,b,c,d; };
    ((H4*)wd)[i] = H4{ (_Float16)(signf(a.x) - signf(b.x)),
                       (_Float16)(signf(a.y) - signf(b.y)),
                       (_Float16)(signf(a.z) - signf(b.z)),
                       (_Float16)(signf(a.w) - signf(b.w)) };
}

__global__ void pack_bdiff(const float* __restrict__ b1, const float* __restrict__ b2,
                           float* __restrict__ bd, int n) {
    int i = blockIdx.x * blockDim.x + threadIdx.x;
    if (i < n) bd[i] = b1[i] - b2[i];
}

// ---------------- WMMA GEMM ----------------
// out[M=N_TOK][N=D_OUT] = Xh[M][K] * Wd[N][K]^T + bd[N]
// block: 256 threads = 8 waves (2 in M x 4 in N), wave tile 64x64 -> block tile 128x256.

__global__ __launch_bounds__(256) void ternary_gemm_wmma(
    const _Float16* __restrict__ X,   // [N_TOK][D_IN] f16
    const _Float16* __restrict__ W,   // [D_OUT][D_IN] f16, values in {-2,0,2}
    const float*    __restrict__ bd,  // [D_OUT]
    float*          __restrict__ out) // [N_TOK][D_OUT]
{
    const int lane = threadIdx.x & 31;
    const int wave = threadIdx.x >> 5;
    const int lm   = lane & 15;   // row (A) / col (B) within 16
    const int lh   = lane >> 4;   // lane-half selects K sub-chunk

    const int mBase = blockIdx.y * 128 + (wave >> 2) * 64;
    const int nBase = blockIdx.x * 256 + (wave & 3) * 64;

    v8f acc[4][4];
#pragma unroll
    for (int i = 0; i < 4; ++i)
#pragma unroll
        for (int j = 0; j < 4; ++j)
            acc[i][j] = (v8f){0.f,0.f,0.f,0.f,0.f,0.f,0.f,0.f};

    // Per-lane base pointers (K-major rows).
    const _Float16* aPtr[4];
    const _Float16* bPtr[4];
#pragma unroll
    for (int i = 0; i < 4; ++i)
        aPtr[i] = X + (size_t)(mBase + i * 16 + lm) * D_IN_ + lh * 8;
#pragma unroll
    for (int j = 0; j < 4; ++j)
        bPtr[j] = W + (size_t)(nBase + j * 16 + lm) * D_IN_ + lh * 8;

    for (int k0 = 0; k0 < D_IN_; k0 += 32) {
        v16h a[4], b[4];
        // A fragment: 16-bit A 16x32 layout — lane m holds row m;
        // lanes 0-15: K 0..7 & 16..23 ; lanes 16-31: K 8..15 & 24..31.
#pragma unroll
        for (int i = 0; i < 4; ++i) {
            union { v16h v; v8h h[2]; } u;
            u.h[0] = *(const v8h*)(aPtr[i] + k0);
            u.h[1] = *(const v8h*)(aPtr[i] + k0 + 16);
            a[i] = u.v;
        }
        // B fragment: 16-bit B 32x16 layout — lane n holds column n;
        // lanes 0-15: K 0..7 & 16..23 ; lanes 16-31: K 8..15 & 24..31.
#pragma unroll
        for (int j = 0; j < 4; ++j) {
            union { v16h v; v8h h[2]; } u;
            u.h[0] = *(const v8h*)(bPtr[j] + k0);
            u.h[1] = *(const v8h*)(bPtr[j] + k0 + 16);
            b[j] = u.v;
        }
        // Prefetch next K slab (global_prefetch_b8).
        if (k0 + 32 < D_IN_) {
            __builtin_prefetch(aPtr[0] + k0 + 32, 0, 1);
            __builtin_prefetch(bPtr[0] + k0 + 32, 0, 1);
        }
#pragma unroll
        for (int i = 0; i < 4; ++i)
#pragma unroll
            for (int j = 0; j < 4; ++j)
                acc[i][j] = __builtin_amdgcn_wmma_f32_16x16x32_f16(
                    /*neg_a=*/false, a[i], /*neg_b=*/false, b[j],
                    /*c_mod=*/(short)0, acc[i][j],
                    /*reuse_a=*/false, /*reuse_b=*/false);
    }

    // Epilogue: C/D layout — VGPR r: lanes 0-15 -> M=r, lanes 16-31 -> M=8+r; N=lane%16.
#pragma unroll
    for (int j = 0; j < 4; ++j) {
        const int col  = nBase + j * 16 + lm;
        const float bj = bd[col];
#pragma unroll
        for (int i = 0; i < 4; ++i) {
#pragma unroll
            for (int r = 0; r < 8; ++r) {
                const int row = mBase + i * 16 + lh * 8 + r;
                out[(size_t)row * D_OUT_ + col] = acc[i][j][r] + bj;
            }
        }
    }
}

// ---------------- launch ----------------

extern "C" void kernel_launch(void* const* d_in, const int* in_sizes, int n_in,
                              void* d_out, int out_size, void* d_ws, size_t ws_size,
                              hipStream_t stream) {
    const float* x  = (const float*)d_in[0];
    const float* W1 = (const float*)d_in[1];
    const float* W2 = (const float*)d_in[2];
    const float* b1 = (const float*)d_in[3];
    const float* b2 = (const float*)d_in[4];
    float* out = (float*)d_out;

    // Workspace layout: x_f16 | wdiff_f16 | bdiff_f32
    _Float16* xh = (_Float16*)d_ws;
    _Float16* wd = xh + (size_t)N_TOK_ * D_IN_;
    float*    bd = (float*)(wd + (size_t)D_OUT_ * D_IN_);

    {
        const int n4 = (N_TOK_ * D_IN_) / 4;
        pack_x_f16<<<(n4 + 255) / 256, 256, 0, stream>>>(x, xh, n4);
    }
    {
        const int n4 = (D_OUT_ * D_IN_) / 4;
        pack_wdiff_f16<<<(n4 + 255) / 256, 256, 0, stream>>>(W1, W2, wd, n4);
    }
    pack_bdiff<<<(D_OUT_ + 255) / 256, 256, 0, stream>>>(b1, b2, bd, D_OUT_);

    dim3 grid(D_OUT_ / 256, N_TOK_ / 128);
    ternary_gemm_wmma<<<grid, 256, 0, stream>>>(xh, wd, bd, out);
    (void)in_sizes; (void)n_in; (void)out_size; (void)ws_size;
}
